// EEGGraphNeuralNetwork_65592740545003
// MI455X (gfx1250) — compile-verified, matrix operand order checked by reading, not run on hardware
//
#include <hip/hip_runtime.h>

typedef __bf16 bf16_t;
typedef bf16_t v16bf __attribute__((ext_vector_type(16)));
typedef bf16_t v8bf  __attribute__((ext_vector_type(8)));
typedef float  v8f   __attribute__((ext_vector_type(8)));

#define N_NODES 19
#define NPAD    32
#define DMODEL  256
#define NHEADS  8
#define HDIM    32
#define PPAIRS  361
#define NLAYERS 2
#define TOK     2          // tokens per workgroup (B-fragment register reuse)

union BF16x16 { v16bf v; v8bf h[2]; };

__device__ inline unsigned short f2bf(float f) {
  unsigned int u = __float_as_uint(f);
  u += 0x7FFFu + ((u >> 16) & 1u);   // round-to-nearest-even
  return (unsigned short)(u >> 16);
}

__device__ inline v8f wmma_bf16(v16bf a, v16bf b, v8f c) {
  return __builtin_amdgcn_wmma_f32_16x16x32_bf16(
      /*neg_a=*/false, a, /*neg_b=*/false, b,
      /*c_mod=*/(short)0, c, /*reuse_a=*/false, /*reuse_b=*/false);
}

// A fragment: 16x32 bf16 tile from LDS (row-major, 256 cols).
// lane m = lane&15, khalf = lane>>4; two contiguous 16B chunks per lane.
__device__ inline v16bf load_fragA(const unsigned short* Abf, int row, int colBase) {
  const bf16_t* p = (const bf16_t*)(Abf + row * DMODEL + colBase);
  BF16x16 u;
  u.h[0] = *(const v8bf*)(p);
  u.h[1] = *(const v8bf*)(p + 16);
  return u.v;
}

// B fragment: 32x16 bf16 tile from global weights stored transposed [n][k].
__device__ inline v16bf load_fragB(const bf16_t* Wt, int n, int colBase) {
  const bf16_t* p = Wt + n * DMODEL + colBase;
  BF16x16 u;
  u.h[0] = *(const v8bf*)(p);
  u.h[1] = *(const v8bf*)(p + 16);
  return u.v;
}

// Two-token GEMM: each B fragment feeds 4 WMMAs (2 tokens x 2 M-tiles).
// 8 waves: wave w owns N-tiles {2w, 2w+1}; K-loop of 8; 8 WMMAs per K-step.
__device__ inline void gemm_proj2(const unsigned short* A0buf, const unsigned short* A1buf,
                                  const bf16_t* Wt, const float* bias,
                                  float* out0, float* out1, int outRows,
                                  int wave, int lane) {
  const int q15 = lane & 15;
  const int kh  = lane >> 4;
  const v8f z = {0.f,0.f,0.f,0.f,0.f,0.f,0.f,0.f};
  v8f a000 = z, a001 = z, a010 = z, a011 = z;   // token0: [mt][nt]
  v8f a100 = z, a101 = z, a110 = z, a111 = z;   // token1
  #pragma unroll
  for (int kt = 0; kt < 8; ++kt) {
    const int cb = kt * 32 + kh * 8;
    v16bf B0  = load_fragB(Wt, wave * 32 + q15, cb);
    v16bf B1  = load_fragB(Wt, wave * 32 + 16 + q15, cb);
    v16bf A00 = load_fragA(A0buf, q15, cb);
    v16bf A01 = load_fragA(A0buf, 16 + q15, cb);
    v16bf A10 = load_fragA(A1buf, q15, cb);
    v16bf A11 = load_fragA(A1buf, 16 + q15, cb);
    a000 = wmma_bf16(A00, B0, a000);
    a001 = wmma_bf16(A00, B1, a001);
    a010 = wmma_bf16(A01, B0, a010);
    a011 = wmma_bf16(A01, B1, a011);
    a100 = wmma_bf16(A10, B0, a100);
    a101 = wmma_bf16(A10, B1, a101);
    a110 = wmma_bf16(A11, B0, a110);
    a111 = wmma_bf16(A11, B1, a111);
  }
  const int n0 = wave * 32 + q15;
  const int n1 = n0 + 16;
  const float b0 = bias[n0];
  const float b1 = bias[n1];
  const int rbase0 = kh * 8;        // D layout: row m = vgpr + 8*khalf
  const int rbase1 = 16 + kh * 8;
  #pragma unroll
  for (int v2 = 0; v2 < 8; ++v2) {
    int r0 = rbase0 + v2, r1 = rbase1 + v2;
    if (r0 < outRows) {
      out0[r0 * DMODEL + n0] = a000[v2] + b0;
      out0[r0 * DMODEL + n1] = a001[v2] + b1;
      out1[r0 * DMODEL + n0] = a100[v2] + b0;
      out1[r0 * DMODEL + n1] = a101[v2] + b1;
    }
    if (r1 < outRows) {
      out0[r1 * DMODEL + n0] = a010[v2] + b0;
      out0[r1 * DMODEL + n1] = a011[v2] + b1;
      out1[r1 * DMODEL + n0] = a110[v2] + b0;
      out1[r1 * DMODEL + n1] = a111[v2] + b1;
    }
  }
}

// Convert 4 weight tensors (L,D,D) f32 -> bf16, transposed to [proj][l][n][k].
__global__ __launch_bounds__(256)
void convert_weights(const float* __restrict__ Wq, const float* __restrict__ Wk,
                     const float* __restrict__ Wv, const float* __restrict__ Wo,
                     unsigned short* __restrict__ out) {
  int idx = blockIdx.x * 256 + threadIdx.x;           // 8 * 65536 total
  if (idx >= 8 * DMODEL * DMODEL) return;
  int pl   = idx >> 16;                               // proj*2 + l
  int rem  = idx & 0xFFFF;
  int n    = rem >> 8;
  int k    = rem & 255;
  int proj = pl >> 1;
  int l    = pl & 1;
  const float* W = (proj == 0) ? Wq : (proj == 1) ? Wk : (proj == 2) ? Wv : Wo;
  out[idx] = f2bf(W[((size_t)l * DMODEL + k) * DMODEL + n]);
}

// LDS budget (bytes):
//   Abf:  2 * 32*256*2            = 32768
//   Ef :  2 * 19*256*4            = 38912
//   qb :  2 * 19*256*4            = 38912
//   kb :  2 * 19*256*4            = 38912   (reused as att@Wo output)
//   vb :  2 * 19*256*4            = 38912
//   sc :  2 * 19*19*8*4           = 23104
//   adj:  19*19*4                 =  1444
//   xs :  2 * 256*4               =  2048
#define SMEM_BYTES (2*(NPAD*DMODEL)*2 + 4*2*(N_NODES*DMODEL)*4 + \
                    2*(N_NODES*N_NODES*NHEADS)*4 + (N_NODES*N_NODES)*4 + 2*DMODEL*4)

__global__ __launch_bounds__(256)
void eeg_gnn_kernel(const float* __restrict__ x, const float* __restrict__ adj,
                    const float* __restrict__ bq, const float* __restrict__ bk,
                    const float* __restrict__ bv, const float* __restrict__ bo,
                    const float* __restrict__ lng, const float* __restrict__ lnb,
                    const unsigned short* __restrict__ WtAll,
                    float* __restrict__ out) {
  extern __shared__ char smem[];
  unsigned short* Abf0 = (unsigned short*)smem;                 // 32x256 bf16
  unsigned short* Abf1 = Abf0 + NPAD * DMODEL;                  // 32x256 bf16
  float* fbase = (float*)(Abf1 + NPAD * DMODEL);
  float* Ef[TOK];  Ef[0] = fbase;                 Ef[1] = Ef[0] + N_NODES * DMODEL;
  float* qb[TOK];  qb[0] = Ef[1] + N_NODES*DMODEL; qb[1] = qb[0] + N_NODES * DMODEL;
  float* kb[TOK];  kb[0] = qb[1] + N_NODES*DMODEL; kb[1] = kb[0] + N_NODES * DMODEL;
  float* vb[TOK];  vb[0] = kb[1] + N_NODES*DMODEL; vb[1] = vb[0] + N_NODES * DMODEL;
  float* sc[TOK];  sc[0] = vb[1] + N_NODES*DMODEL; sc[1] = sc[0] + N_NODES*N_NODES*NHEADS;
  float* adjs = sc[1] + N_NODES * N_NODES * NHEADS;             // 19x19
  float* xs[TOK]; xs[0] = adjs + N_NODES * N_NODES; xs[1] = xs[0] + DMODEL;
  unsigned short* Abf[TOK] = {Abf0, Abf1};

  const int tid  = threadIdx.x;
  const int lane = tid & 31;
  const int wave = tid >> 5;
  const int bp0  = blockIdx.x * TOK;          // 2 tokens per block
  int piA[TOK], pjA[TOK];
  #pragma unroll
  for (int t = 0; t < TOK; ++t) {
    int p = (bp0 + t) % PPAIRS;
    piA[t] = p / N_NODES;
    pjA[t] = p % N_NODES;
  }

  // load token vectors + adjacency
  #pragma unroll
  for (int t = 0; t < TOK; ++t)
    xs[t][tid] = x[(size_t)(bp0 + t) * DMODEL + tid];
  for (int i = tid; i < N_NODES * N_NODES; i += 256) adjs[i] = adj[i];
  __syncthreads();

  // E init: row n = ((n==pi)+(n==pj)) * x
  #pragma unroll
  for (int t = 0; t < TOK; ++t)
    for (int i = tid; i < N_NODES * DMODEL; i += 256) {
      int r = i >> 8, d = i & 255;
      float c = (float)((r == piA[t]) + (r == pjA[t]));
      Ef[t][i] = c * xs[t][d];
    }
  __syncthreads();

  const float inv_sqrt_hd = 0.17677669529663689f;  // 1/sqrt(32)

  for (int l = 0; l < NLAYERS; ++l) {
    const bf16_t* Wtq = (const bf16_t*)(WtAll + (size_t)(0 * 2 + l) * DMODEL * DMODEL);
    const bf16_t* Wtk = (const bf16_t*)(WtAll + (size_t)(1 * 2 + l) * DMODEL * DMODEL);
    const bf16_t* Wtv = (const bf16_t*)(WtAll + (size_t)(2 * 2 + l) * DMODEL * DMODEL);
    const bf16_t* Wto = (const bf16_t*)(WtAll + (size_t)(3 * 2 + l) * DMODEL * DMODEL);

    // stage E -> bf16 A buffers (zero-pad rows >= 19)
    #pragma unroll
    for (int t = 0; t < TOK; ++t)
      for (int i = tid; i < NPAD * DMODEL; i += 256) {
        int r = i >> 8;
        Abf[t][i] = f2bf(r < N_NODES ? Ef[t][i] : 0.f);
      }
    __syncthreads();

    gemm_proj2(Abf0, Abf1, Wtq, bq + l * DMODEL, qb[0], qb[1], N_NODES, wave, lane);
    gemm_proj2(Abf0, Abf1, Wtk, bk + l * DMODEL, kb[0], kb[1], N_NODES, wave, lane);
    gemm_proj2(Abf0, Abf1, Wtv, bv + l * DMODEL, vb[0], vb[1], N_NODES, wave, lane);
    __syncthreads();

    // scores[n][m][h] = (q_nh . k_mh)/sqrt(hd), masked by adjacency
    #pragma unroll
    for (int t = 0; t < TOK; ++t)
      for (int i = tid; i < N_NODES * N_NODES * NHEADS; i += 256) {
        int h  = i & 7;
        int nm = i >> 3;
        int mm = nm % N_NODES;
        int nn = nm / N_NODES;
        float s;
        if (adjs[nn * N_NODES + mm] > 0.f) {
          const float* qp = qb[t] + nn * DMODEL + h * HDIM;
          const float* kp = kb[t] + mm * DMODEL + h * HDIM;
          float acc = 0.f;
          #pragma unroll 8
          for (int e = 0; e < HDIM; ++e) acc += qp[e] * kp[e];
          s = acc * inv_sqrt_hd;
        } else {
          s = -__builtin_inff();
        }
        sc[t][i] = s;
      }
    __syncthreads();

    // softmax over m per (n,h)
    #pragma unroll
    for (int t = 0; t < TOK; ++t)
      for (int i = tid; i < N_NODES * NHEADS; i += 256) {
        int nn = i >> 3, h = i & 7;
        float mx = -__builtin_inff();
        for (int mm = 0; mm < N_NODES; ++mm)
          mx = fmaxf(mx, sc[t][(nn * N_NODES + mm) * NHEADS + h]);
        float sum = 0.f;
        for (int mm = 0; mm < N_NODES; ++mm) {
          int ix = (nn * N_NODES + mm) * NHEADS + h;
          float e = __expf(sc[t][ix] - mx);
          sc[t][ix] = e;
          sum += e;
        }
        float inv = (sum > 0.f) ? (1.f / sum) : 0.f;
        for (int mm = 0; mm < N_NODES; ++mm)
          sc[t][(nn * N_NODES + mm) * NHEADS + h] *= inv;
      }
    __syncthreads();

    // att = w @ v  (reuse qb as att buffer; q dead after scores)
    #pragma unroll
    for (int t = 0; t < TOK; ++t)
      for (int i = tid; i < N_NODES * DMODEL; i += 256) {
        int nn = i >> 8, d = i & 255, h = d >> 5;
        float acc = 0.f;
        for (int mm = 0; mm < N_NODES; ++mm)
          acc += sc[t][(nn * N_NODES + mm) * NHEADS + h] * vb[t][mm * DMODEL + d];
        qb[t][i] = acc;
      }
    __syncthreads();

    // stage att -> bf16 A buffers (zero-pad rows >= 19)
    #pragma unroll
    for (int t = 0; t < TOK; ++t)
      for (int i = tid; i < NPAD * DMODEL; i += 256) {
        int r = i >> 8;
        Abf[t][i] = f2bf(r < N_NODES ? qb[t][i] : 0.f);
      }
    __syncthreads();

    // att @ Wo + bo -> kb (k dead after scores)
    gemm_proj2(Abf0, Abf1, Wto, bo + l * DMODEL, kb[0], kb[1], N_NODES, wave, lane);
    __syncthreads();

    // layernorm(att@Wo + bo + E) per row, wave32 shuffle reduction
    const float* g  = lng + l * DMODEL;
    const float* bb = lnb + l * DMODEL;
    #pragma unroll
    for (int t = 0; t < TOK; ++t)
      for (int r = wave; r < N_NODES; r += 8) {
        float vals[8];
        float s = 0.f, s2 = 0.f;
        #pragma unroll
        for (int i2 = 0; i2 < 8; ++i2) {
          int d = lane + i2 * 32;
          float xv = kb[t][r * DMODEL + d] + Ef[t][r * DMODEL + d];
          vals[i2] = xv; s += xv; s2 += xv * xv;
        }
        #pragma unroll
        for (int off = 16; off > 0; off >>= 1) {
          s  += __shfl_xor(s,  off);
          s2 += __shfl_xor(s2, off);
        }
        float mu  = s * (1.f / DMODEL);
        float var = s2 * (1.f / DMODEL) - mu * mu;
        float inv = rsqrtf(var + 1e-5f);
        #pragma unroll
        for (int i2 = 0; i2 < 8; ++i2) {
          int d = lane + i2 * 32;
          Ef[t][r * DMODEL + d] = (vals[i2] - mu) * inv * g[d] + bb[d];
        }
      }
    __syncthreads();
  }

  // readout: 0.5*(E[pi] + E[pj]) + x
  #pragma unroll
  for (int t = 0; t < TOK; ++t)
    for (int d = tid; d < DMODEL; d += 256) {
      out[(size_t)(bp0 + t) * DMODEL + d] =
          0.5f * (Ef[t][piA[t] * DMODEL + d] + Ef[t][pjA[t] * DMODEL + d]) + xs[t][d];
    }
}

extern "C" void kernel_launch(void* const* d_in, const int* in_sizes, int n_in,
                              void* d_out, int out_size, void* d_ws, size_t ws_size,
                              hipStream_t stream) {
  const float* x   = (const float*)d_in[0];
  const float* adj = (const float*)d_in[1];
  const float* Wq  = (const float*)d_in[2];
  const float* bq  = (const float*)d_in[3];
  const float* Wk  = (const float*)d_in[4];
  const float* bk  = (const float*)d_in[5];
  const float* Wv  = (const float*)d_in[6];
  const float* bv  = (const float*)d_in[7];
  const float* Wo  = (const float*)d_in[8];
  const float* bo  = (const float*)d_in[9];
  const float* lng = (const float*)d_in[10];
  const float* lnb = (const float*)d_in[11];
  unsigned short* Wt = (unsigned short*)d_ws;   // 8 * 65536 bf16 = 1 MB
  float* out = (float*)d_out;

  convert_weights<<<(8 * DMODEL * DMODEL + 255) / 256, 256, 0, stream>>>(Wq, Wk, Wv, Wo, Wt);

  const int nBlocks = (16 * PPAIRS) / TOK;  // 5776 / 2 = 2888
  eeg_gnn_kernel<<<nBlocks, 256, SMEM_BYTES, stream>>>(
      x, adj, bq, bk, bv, bo, lng, lnb, Wt, out);
}
